// EncodeProcessDecodeGNS_9337258902441
// MI455X (gfx1250) — compile-verified, matrix operand order checked by reading, not run on hardware
//
#include <hip/hip_runtime.h>
#include <hip/hip_bf16.h>
#include <math.h>
#include <stdint.h>

// ---------------------------------------------------------------------------
// EncodeProcessDecode GNS for MI455X (gfx1250, wave32, WMMA bf16)
//  - edge MLP: 32-edge blocks, each wave owns 32 rows x 32 cols (4 WMMA accs)
//    -> halves B-matrix (weight) L2 traffic vs 16-row blocks
//  - X tile staged with global_load_async_to_lds_b128 (pure byte copy:
//    h mirrored in bf16, e-residual scale 2^l folded into W1 rows 512..767)
// ---------------------------------------------------------------------------

typedef __bf16 bf16;
typedef __attribute__((ext_vector_type(16))) __bf16 v16bf;
typedef __attribute__((ext_vector_type(8)))  float  v8f;
typedef __attribute__((ext_vector_type(4)))  float  f32x4;

__device__ __forceinline__ v16bf load_frag(const bf16* p0, int pitch) {
  int lane = threadIdx.x & 31;
  const bf16* p = p0 + (lane & 15) * pitch + (lane >> 4) * 8;
  v16bf f;
  ((f32x4*)&f)[0] = *(const f32x4*)(p);
  ((f32x4*)&f)[1] = *(const f32x4*)(p + 16);
  return f;
}

#define WMMA_BF16(A, B, C) \
  __builtin_amdgcn_wmma_f32_16x16x32_bf16(false, (A), false, (B), (short)0, (C), false, false)

__device__ __forceinline__ float gelu_erf(float x) {
  return 0.5f * x * (1.0f + erff(x * 0.7071067811865476f));
}

__device__ __forceinline__ void epi_bf16(bf16* Y, int pitch, const v8f& acc, int n0,
                                         const float* bias, bool act, int rows_valid) {
  int lane = threadIdx.x & 31;
  int col  = n0 + (lane & 15);
  int r0   = (lane >> 4) * 8;
  float b  = bias[col];
#pragma unroll
  for (int v = 0; v < 8; ++v) {
    int r = r0 + v;
    if (r < rows_valid) {
      float t = acc[v] + b;
      if (act) t = gelu_erf(t);
      Y[(size_t)r * pitch + col] = (bf16)t;
    }
  }
}

__device__ __forceinline__ void epi_f32(float* Y, int pitch, const v8f& acc, int n0,
                                        const float* bias, bool act, int rows_valid) {
  int lane = threadIdx.x & 31;
  int col  = n0 + (lane & 15);
  int r0   = (lane >> 4) * 8;
  float b  = bias[col];
#pragma unroll
  for (int v = 0; v < 8; ++v) {
    int r = r0 + v;
    if (r < rows_valid) {
      float t = acc[v] + b;
      if (act) t = gelu_erf(t);
      Y[(size_t)r * pitch + col] = t;
    }
  }
}

__device__ __forceinline__ void epi_pair(float* F, bf16* B, int pitch, const v8f& acc,
                                         int n0, const float* bias, int rows_valid) {
  int lane = threadIdx.x & 31;
  int col  = n0 + (lane & 15);
  int r0   = (lane >> 4) * 8;
  float b  = bias[col];
#pragma unroll
  for (int v = 0; v < 8; ++v) {
    int r = r0 + v;
    if (r < rows_valid) {
      float t = acc[v] + b;
      F[(size_t)r * pitch + col] = t;
      B[(size_t)r * pitch + col] = (bf16)t;
    }
  }
}

// ---- 16-row block GEMM: 8 waves, wave w -> cols [32w,32w+32) ---------------
__device__ __forceinline__ void gemm16_bf16(const bf16* Xl, int pitchX, int K,
                                            const bf16* __restrict__ Wt,
                                            const float* __restrict__ bias,
                                            bf16* Yl, int pitchY, bool act, int rows_valid) {
  int n0 = (threadIdx.x >> 5) * 32;
  v8f a0 = {}, a1 = {};
  for (int k = 0; k < K; k += 32) {
    v16bf a  = load_frag(Xl + k, pitchX);
    v16bf b0 = load_frag(Wt + (size_t)n0 * K + k, K);
    v16bf b1 = load_frag(Wt + (size_t)(n0 + 16) * K + k, K);
    a0 = WMMA_BF16(a, b0, a0);
    a1 = WMMA_BF16(a, b1, a1);
  }
  epi_bf16(Yl, pitchY, a0, n0,      bias, act, rows_valid);
  epi_bf16(Yl, pitchY, a1, n0 + 16, bias, act, rows_valid);
}

__device__ __forceinline__ void gemm16_f32(const bf16* Xl, int pitchX, int K,
                                           const bf16* __restrict__ Wt,
                                           const float* __restrict__ bias,
                                           float* Yl, int pitchY, bool act, int rows_valid) {
  int n0 = (threadIdx.x >> 5) * 32;
  v8f a0 = {}, a1 = {};
  for (int k = 0; k < K; k += 32) {
    v16bf a  = load_frag(Xl + k, pitchX);
    v16bf b0 = load_frag(Wt + (size_t)n0 * K + k, K);
    v16bf b1 = load_frag(Wt + (size_t)(n0 + 16) * K + k, K);
    a0 = WMMA_BF16(a, b0, a0);
    a1 = WMMA_BF16(a, b1, a1);
  }
  epi_f32(Yl, pitchY, a0, n0,      bias, act, rows_valid);
  epi_f32(Yl, pitchY, a1, n0 + 16, bias, act, rows_valid);
}

__device__ __forceinline__ void gemm16_hpair(const bf16* Xl, int pitchX, int K,
                                             const bf16* __restrict__ Wt,
                                             const float* __restrict__ bias,
                                             float* F, bf16* B, int pitchY, int rows_valid) {
  int n0 = (threadIdx.x >> 5) * 32;
  v8f a0 = {}, a1 = {};
  for (int k = 0; k < K; k += 32) {
    v16bf a  = load_frag(Xl + k, pitchX);
    v16bf b0 = load_frag(Wt + (size_t)n0 * K + k, K);
    v16bf b1 = load_frag(Wt + (size_t)(n0 + 16) * K + k, K);
    a0 = WMMA_BF16(a, b0, a0);
    a1 = WMMA_BF16(a, b1, a1);
  }
  epi_pair(F, B, pitchY, a0, n0,      bias, rows_valid);
  epi_pair(F, B, pitchY, a1, n0 + 16, bias, rows_valid);
}

// ---- 32-row block GEMM: wave owns all 32 rows x its 32 cols ----------------
__device__ __forceinline__ void gemm32_bf16(const bf16* Xl, int pitchX, int K,
                                            const bf16* __restrict__ Wt,
                                            const float* __restrict__ bias,
                                            bf16* Yl, int pitchY, bool act) {
  int n0 = (threadIdx.x >> 5) * 32;
  v8f a00 = {}, a01 = {}, a10 = {}, a11 = {};
  for (int k = 0; k < K; k += 32) {
    v16bf A0 = load_frag(Xl + k, pitchX);
    v16bf A1 = load_frag(Xl + 16 * pitchX + k, pitchX);
    v16bf B0 = load_frag(Wt + (size_t)n0 * K + k, K);
    v16bf B1 = load_frag(Wt + (size_t)(n0 + 16) * K + k, K);
    a00 = WMMA_BF16(A0, B0, a00);
    a01 = WMMA_BF16(A0, B1, a01);
    a10 = WMMA_BF16(A1, B0, a10);
    a11 = WMMA_BF16(A1, B1, a11);
  }
  epi_bf16(Yl,               pitchY, a00, n0,      bias, act, 16);
  epi_bf16(Yl,               pitchY, a01, n0 + 16, bias, act, 16);
  epi_bf16(Yl + 16 * pitchY, pitchY, a10, n0,      bias, act, 16);
  epi_bf16(Yl + 16 * pitchY, pitchY, a11, n0 + 16, bias, act, 16);
}

__device__ __forceinline__ void gemm32_f32(const bf16* Xl, int pitchX, int K,
                                           const bf16* __restrict__ Wt,
                                           const float* __restrict__ bias,
                                           float* Yl, int pitchY, bool act) {
  int n0 = (threadIdx.x >> 5) * 32;
  v8f a00 = {}, a01 = {}, a10 = {}, a11 = {};
  for (int k = 0; k < K; k += 32) {
    v16bf A0 = load_frag(Xl + k, pitchX);
    v16bf A1 = load_frag(Xl + 16 * pitchX + k, pitchX);
    v16bf B0 = load_frag(Wt + (size_t)n0 * K + k, K);
    v16bf B1 = load_frag(Wt + (size_t)(n0 + 16) * K + k, K);
    a00 = WMMA_BF16(A0, B0, a00);
    a01 = WMMA_BF16(A0, B1, a01);
    a10 = WMMA_BF16(A1, B0, a10);
    a11 = WMMA_BF16(A1, B1, a11);
  }
  epi_f32(Yl,               pitchY, a00, n0,      bias, act, 16);
  epi_f32(Yl,               pitchY, a01, n0 + 16, bias, act, 16);
  epi_f32(Yl + 16 * pitchY, pitchY, a10, n0,      bias, act, 16);
  epi_f32(Yl + 16 * pitchY, pitchY, a11, n0 + 16, bias, act, 16);
}

// ---- weight prep: f32 [K][N] -> bf16 transposed [N][K]; rows k>=kstart
// multiplied by `scale` (folds e_l = 2^l * e0 into W1) -----------------------
__global__ void k_transpose(const float* __restrict__ W, bf16* __restrict__ Wt,
                            int K, int Nc, int kstart, float scale) {
  int idx = blockIdx.x * blockDim.x + threadIdx.x;
  if (idx < K * Nc) {
    int k = idx / Nc, n = idx % Nc;
    float v = W[idx];
    if (k >= kstart) v *= scale;
    Wt[(size_t)n * K + k] = (bf16)v;
  }
}

__global__ void k_degree(const int* __restrict__ ei, float* __restrict__ cnt, int Etot) {
  int e = blockIdx.x * blockDim.x + threadIdx.x;
  if (e < Etot) atomicAdd(&cnt[ei[Etot + e]], 1.0f);
}

// ---- node encoder ----------------------------------------------------------
__global__ __launch_bounds__(256) void k_node_enc(
    const float* __restrict__ x, const float* __restrict__ w1, const float* __restrict__ b1,
    const bf16* __restrict__ w2t, const float* __restrict__ b2,
    float* __restrict__ h, bf16* __restrict__ hb, int Ntot) {
  __shared__ float Xf[16 * 16];
  __shared__ __align__(16) bf16 Y[16 * 264];
  int nbase = blockIdx.x * 16;
  int tid = threadIdx.x;
  {
    int r = tid >> 4, k = tid & 15;
    int n = min(nbase + r, Ntot - 1);
    Xf[tid] = x[(size_t)n * 16 + k];
  }
  __syncthreads();
  {
    int col = tid;
    for (int r = 0; r < 16; ++r) {
      float a = b1[col];
#pragma unroll
      for (int k = 0; k < 16; ++k) a += Xf[r * 16 + k] * w1[k * 256 + col];
      Y[r * 264 + col] = (bf16)gelu_erf(a);
    }
  }
  __syncthreads();
  int rows = min(16, Ntot - nbase);
  gemm16_hpair(Y, 264, 256, w2t, b2,
               h + (size_t)nbase * 256, hb + (size_t)nbase * 256, 256, rows);
}

// ---- edge encoder ----------------------------------------------------------
__global__ __launch_bounds__(256) void k_edge_enc(
    const float* __restrict__ ea,
    const bf16* __restrict__ w1t, const float* __restrict__ b1,
    const bf16* __restrict__ w2t, const float* __restrict__ b2,
    bf16* __restrict__ e0, int Etot) {
  __shared__ __align__(16) bf16 X[16 * 264];
  __shared__ __align__(16) bf16 Y[16 * 264];
  int ebase = blockIdx.x * 16;
  int c = threadIdx.x;
  for (int r = 0; r < 16; ++r) {
    int e = min(ebase + r, Etot - 1);
    X[r * 264 + c] = (bf16)ea[(size_t)e * 256 + c];
  }
  __syncthreads();
  gemm16_bf16(X, 264, 256, w1t, b1, Y, 264, true, 16);
  __syncthreads();
  int rows = min(16, Etot - ebase);
  gemm16_bf16(Y, 264, 256, w2t, b2, e0 + (size_t)ebase * 256, 256, false, rows);
}

// ---- fused edge MLP + LayerNorm + scatter-add (32 edges / block) -----------
__global__ __launch_bounds__(256) void k_edge_mlp(
    const bf16* __restrict__ hb, const bf16* __restrict__ e0,
    const int* __restrict__ ei, int Etot,
    const bf16* __restrict__ w1t, const float* __restrict__ b1,
    const bf16* __restrict__ w2t, const float* __restrict__ b2,
    const bf16* __restrict__ w3t, const float* __restrict__ b3,
    const float* __restrict__ lng, const float* __restrict__ lnb,
    float* __restrict__ s) {
  __shared__ __align__(16) bf16 X[32 * 776];
  __shared__ __align__(16) bf16 Y[32 * 264];
  __shared__ float Z[32 * 256];
  __shared__ int dsts[32], srcs[32];

  int ebase = blockIdx.x * 32;
  int tid = threadIdx.x;
  int lane = tid & 31, wv = tid >> 5;

  if (tid < 32) {
    int e = min(ebase + tid, Etot - 1);
    srcs[tid] = ei[e];
    dsts[tid] = ei[Etot + e];
  }
  __syncthreads();

  // async-stage X: 96 pure 512-byte row-part copies, 16B per lane
  uint32_t xoff = (uint32_t)(uintptr_t)(&X[0]);
  for (int pc = wv; pc < 96; pc += 8) {
    int part = pc >> 5;       // 0: h[dst], 1: h[src], 2: e0
    int r    = pc & 31;
    const bf16* gsrc;
    if (part == 0)      gsrc = hb + (size_t)dsts[r] * 256;
    else if (part == 1) gsrc = hb + (size_t)srcs[r] * 256;
    else                gsrc = e0 + (size_t)min(ebase + r, Etot - 1) * 256;
    uint64_t ga = (uint64_t)(uintptr_t)(gsrc + (lane << 3));
    uint32_t la = xoff + (uint32_t)((r * 776 + (part << 8) + (lane << 3)) << 1);
    asm volatile("global_load_async_to_lds_b128 %0, %1, off"
                 :: "v"(la), "v"(ga) : "memory");
  }
  asm volatile("s_wait_asynccnt 0" ::: "memory");
  __syncthreads();

  gemm32_bf16(X, 776, 768, w1t, b1, Y, 264, true);
  __syncthreads();
  gemm32_bf16(Y, 264, 256, w2t, b2, X, 264, true);   // Y2 reuses X region
  __syncthreads();
  gemm32_f32(X, 264, 256, w3t, b3, Z, 256, false);
  __syncthreads();

  for (int rr = 0; rr < 4; ++rr) {
    int r = wv * 4 + rr;
    float vals[8];
    float sum = 0.f;
#pragma unroll
    for (int j = 0; j < 8; ++j) { vals[j] = Z[r * 256 + lane + j * 32]; sum += vals[j]; }
#pragma unroll
    for (int o = 16; o > 0; o >>= 1) sum += __shfl_down(sum, o, 32);
    float mu = __shfl(sum, 0, 32) * (1.f / 256.f);
    float sq = 0.f;
#pragma unroll
    for (int j = 0; j < 8; ++j) { float d = vals[j] - mu; sq += d * d; }
#pragma unroll
    for (int o = 16; o > 0; o >>= 1) sq += __shfl_down(sq, o, 32);
    float rstd = rsqrtf(__shfl(sq, 0, 32) * (1.f / 256.f) + 1e-5f);
    if (ebase + r < Etot) {
      size_t drow = (size_t)dsts[r] * 256;
#pragma unroll
      for (int j = 0; j < 8; ++j) {
        int cc = lane + j * 32;
        float o2 = (vals[j] - mu) * rstd * lng[cc] + lnb[cc];
        atomicAdd(&s[drow + cc], o2);
      }
    }
  }
}

// ---- fused node MLP + LayerNorm + residual ---------------------------------
__global__ __launch_bounds__(256) void k_node_mlp(
    float* __restrict__ h, bf16* __restrict__ hb,
    const float* __restrict__ s, const float* __restrict__ cnt, int Ntot,
    const bf16* __restrict__ w1t, const float* __restrict__ b1,
    const bf16* __restrict__ w2t, const float* __restrict__ b2,
    const bf16* __restrict__ w3t, const float* __restrict__ b3,
    const float* __restrict__ lng, const float* __restrict__ lnb) {
  __shared__ __align__(16) bf16 X[16 * 520];
  __shared__ __align__(16) bf16 Y[16 * 264];
  __shared__ float Z[16 * 256];

  int nbase = blockIdx.x * 16;
  int c = threadIdx.x;
  for (int r = 0; r < 16; ++r) {
    int n = min(nbase + r, Ntot - 1);
    X[r * 520 + c] = hb[(size_t)n * 256 + c];
    float cn = fmaxf(cnt[n], 1.f);
    X[r * 520 + 256 + c] = (bf16)(s[(size_t)n * 256 + c] / cn);
  }
  __syncthreads();

  gemm16_bf16(X, 520, 512, w1t, b1, Y, 264, true, 16);
  __syncthreads();
  gemm16_bf16(Y, 264, 256, w2t, b2, X, 264, true, 16);
  __syncthreads();
  gemm16_f32(X, 264, 256, w3t, b3, Z, 256, false, 16);
  __syncthreads();

  int wv = threadIdx.x >> 5, lane = threadIdx.x & 31;
  for (int rr = 0; rr < 2; ++rr) {
    int r = wv * 2 + rr;
    float vals[8];
    float sum = 0.f;
#pragma unroll
    for (int j = 0; j < 8; ++j) { vals[j] = Z[r * 256 + lane + j * 32]; sum += vals[j]; }
#pragma unroll
    for (int o = 16; o > 0; o >>= 1) sum += __shfl_down(sum, o, 32);
    float mu = __shfl(sum, 0, 32) * (1.f / 256.f);
    float sq = 0.f;
#pragma unroll
    for (int j = 0; j < 8; ++j) { float d = vals[j] - mu; sq += d * d; }
#pragma unroll
    for (int o = 16; o > 0; o >>= 1) sq += __shfl_down(sq, o, 32);
    float rstd = rsqrtf(__shfl(sq, 0, 32) * (1.f / 256.f) + 1e-5f);
    if (nbase + r < Ntot) {
      size_t row = (size_t)(nbase + r) * 256;
#pragma unroll
      for (int j = 0; j < 8; ++j) {
        int cc = lane + j * 32;
        float u = (vals[j] - mu) * rstd * lng[cc] + lnb[cc];
        float t = h[row + cc] + u;
        h[row + cc]  = t;
        hb[row + cc] = (bf16)t;
      }
    }
  }
}

// ---- decoder ---------------------------------------------------------------
__global__ __launch_bounds__(256) void k_decoder(
    const bf16* __restrict__ hb,
    const bf16* __restrict__ w1t, const float* __restrict__ b1,
    const float* __restrict__ w2, const float* __restrict__ b2,
    float* __restrict__ out, int Ntot) {
  __shared__ __align__(16) bf16 X[16 * 264];
  __shared__ __align__(16) bf16 Y[16 * 264];
  int nbase = blockIdx.x * 16;
  int c = threadIdx.x;
  for (int r = 0; r < 16; ++r) {
    int n = min(nbase + r, Ntot - 1);
    X[r * 264 + c] = hb[(size_t)n * 256 + c];
  }
  __syncthreads();
  gemm16_bf16(X, 264, 256, w1t, b1, Y, 264, true, 16);
  __syncthreads();
  if (c < 48) {
    int r = c / 3, o = c % 3;
    if (nbase + r < Ntot) {
      float acc = b2[o];
      for (int k = 0; k < 256; ++k) acc += (float)Y[r * 264 + k] * w2[k * 3 + o];
      out[(size_t)(nbase + r) * 3 + o] = acc;
    }
  }
}

// ---------------------------------------------------------------------------
extern "C" void kernel_launch(void* const* d_in, const int* in_sizes, int n_in,
                              void* d_out, int out_size, void* d_ws, size_t ws_size,
                              hipStream_t stream) {
  (void)n_in; (void)out_size; (void)ws_size;
  const float* x        = (const float*)d_in[0];
  const int*   ei       = (const int*)  d_in[1];
  const float* ea       = (const float*)d_in[2];
  const float* enc_n_w1 = (const float*)d_in[3];
  const float* enc_n_b1 = (const float*)d_in[4];
  const float* enc_n_w2 = (const float*)d_in[5];
  const float* enc_n_b2 = (const float*)d_in[6];
  const float* enc_e_w1 = (const float*)d_in[7];
  const float* enc_e_b1 = (const float*)d_in[8];
  const float* enc_e_w2 = (const float*)d_in[9];
  const float* enc_e_b2 = (const float*)d_in[10];
  const float* em_w1 = (const float*)d_in[11];
  const float* em_b1 = (const float*)d_in[12];
  const float* em_w2 = (const float*)d_in[13];
  const float* em_b2 = (const float*)d_in[14];
  const float* em_w3 = (const float*)d_in[15];
  const float* em_b3 = (const float*)d_in[16];
  const float* em_lng = (const float*)d_in[17];
  const float* em_lnb = (const float*)d_in[18];
  const float* nm_w1 = (const float*)d_in[19];
  const float* nm_b1 = (const float*)d_in[20];
  const float* nm_w2 = (const float*)d_in[21];
  const float* nm_b2 = (const float*)d_in[22];
  const float* nm_w3 = (const float*)d_in[23];
  const float* nm_b3 = (const float*)d_in[24];
  const float* nm_lng = (const float*)d_in[25];
  const float* nm_lnb = (const float*)d_in[26];
  const float* dec_w1 = (const float*)d_in[27];
  const float* dec_b1 = (const float*)d_in[28];
  const float* dec_w2 = (const float*)d_in[29];
  const float* dec_b2 = (const float*)d_in[30];

  const int N = in_sizes[0] / 16;
  const int E = in_sizes[2] / 256;
  const int M = 6;

  char* p = (char*)d_ws;
  auto carve = [&](size_t bytes) -> char* {
    char* r = p; p += (bytes + 255) & ~(size_t)255; return r;
  };
  float* h    = (float*)carve((size_t)N * 256 * 4);
  bf16*  hb   = (bf16*) carve((size_t)N * 256 * 2);
  float* s    = (float*)carve((size_t)N * 256 * 4);
  float* cnt  = (float*)carve((size_t)N * 4);
  bf16*  e0   = (bf16*) carve((size_t)E * 256 * 2);
  bf16* enc_n_w2t = (bf16*)carve(256 * 256 * 2);
  bf16* enc_e_w1t = (bf16*)carve(256 * 256 * 2);
  bf16* enc_e_w2t = (bf16*)carve(256 * 256 * 2);
  bf16* dec_w1t   = (bf16*)carve(256 * 256 * 2);
  bf16* em_w1t = (bf16*)carve((size_t)M * 768 * 256 * 2);
  bf16* em_w2t = (bf16*)carve((size_t)M * 256 * 256 * 2);
  bf16* em_w3t = (bf16*)carve((size_t)M * 256 * 256 * 2);
  bf16* nm_w1t = (bf16*)carve((size_t)M * 512 * 256 * 2);
  bf16* nm_w2t = (bf16*)carve((size_t)M * 256 * 256 * 2);
  bf16* nm_w3t = (bf16*)carve((size_t)M * 256 * 256 * 2);

  auto tr = [&](const float* W, bf16* Wt, int K, int Nc, int kstart, float scale) {
    int tot = K * Nc;
    k_transpose<<<(tot + 255) / 256, 256, 0, stream>>>(W, Wt, K, Nc, kstart, scale);
  };
  tr(enc_n_w2, enc_n_w2t, 256, 256, 256, 1.f);
  tr(enc_e_w1, enc_e_w1t, 256, 256, 256, 1.f);
  tr(enc_e_w2, enc_e_w2t, 256, 256, 256, 1.f);
  tr(dec_w1,   dec_w1t,   256, 256, 256, 1.f);
  for (int l = 0; l < M; ++l) {
    tr(em_w1 + (size_t)l * 768 * 256, em_w1t + (size_t)l * 768 * 256, 768, 256,
       512, (float)(1u << l));                       // fold e_l = 2^l * e0
    tr(em_w2 + (size_t)l * 256 * 256, em_w2t + (size_t)l * 256 * 256, 256, 256, 256, 1.f);
    tr(em_w3 + (size_t)l * 256 * 256, em_w3t + (size_t)l * 256 * 256, 256, 256, 256, 1.f);
    tr(nm_w1 + (size_t)l * 512 * 256, nm_w1t + (size_t)l * 512 * 256, 512, 256, 512, 1.f);
    tr(nm_w2 + (size_t)l * 256 * 256, nm_w2t + (size_t)l * 256 * 256, 256, 256, 256, 1.f);
    tr(nm_w3 + (size_t)l * 256 * 256, nm_w3t + (size_t)l * 256 * 256, 256, 256, 256, 1.f);
  }

  int nodeBlocks   = (N + 15) / 16;   // 625
  int edgeBlocks16 = (E + 15) / 16;   // 18750 (encoder)
  int edgeBlocks32 = (E + 31) / 32;   // 9375  (edge MLP)

  k_node_enc<<<nodeBlocks, 256, 0, stream>>>(x, enc_n_w1, enc_n_b1, enc_n_w2t, enc_n_b2,
                                             h, hb, N);
  k_edge_enc<<<edgeBlocks16, 256, 0, stream>>>(ea, enc_e_w1t, enc_e_b1, enc_e_w2t,
                                               enc_e_b2, e0, E);
  hipMemsetAsync(cnt, 0, (size_t)N * 4, stream);
  k_degree<<<(E + 255) / 256, 256, 0, stream>>>(ei, cnt, E);

  for (int l = 0; l < M; ++l) {
    hipMemsetAsync(s, 0, (size_t)N * 256 * 4, stream);
    k_edge_mlp<<<edgeBlocks32, 256, 0, stream>>>(
        hb, e0, ei, E,
        em_w1t + (size_t)l * 768 * 256, em_b1 + l * 256,
        em_w2t + (size_t)l * 256 * 256, em_b2 + l * 256,
        em_w3t + (size_t)l * 256 * 256, em_b3 + l * 256,
        em_lng + l * 256, em_lnb + l * 256, s);
    k_node_mlp<<<nodeBlocks, 256, 0, stream>>>(
        h, hb, s, cnt, N,
        nm_w1t + (size_t)l * 512 * 256, nm_b1 + l * 256,
        nm_w2t + (size_t)l * 256 * 256, nm_b2 + l * 256,
        nm_w3t + (size_t)l * 256 * 256, nm_b3 + l * 256,
        nm_lng + l * 256, nm_lnb + l * 256);
  }

  k_decoder<<<nodeBlocks, 256, 0, stream>>>(hb, dec_w1t, dec_b1, dec_w2, dec_b2,
                                            (float*)d_out, N);
}
